// MyRnn_49168785605488
// MI455X (gfx1250) — matrix operand order repaired and checked by reading, not compiled, not run on hardware
//
#include <hip/hip_runtime.h>
#include <hip/hip_bf16.h>

// ---------------------------------------------------------------------------
// 2-layer LSTM (B=1024, T=80, E=100, U=512) for MI455X (gfx1250, wave32).
//
// Strategy:
//  - Per timestep, layer L computes z = [x|h] @ [W;U]^T-major + b with a
//    single f16 WMMA GEMM (f32 accumulate), then an elementwise LSTM update.
//  - Weights are repacked once per call into K-major f16 ("Bt"), padded so
//    K is a multiple of 32 (E: 100 -> 128).
//  - Activations kept as f16 copies inside concatenated A panels; cell state
//    and gate math in f32.
// ---------------------------------------------------------------------------

#define BATCH 1024
#define SEQ   80
#define EMB   100
#define EMBP  128          // padded embedding (mult of 32)
#define UNITS 512
#define GATES 2048         // 4*UNITS
#define K1    (EMBP + UNITS)   // 640, layer-1 GEMM K
#define K2    (UNITS + UNITS)  // 1024, layer-2 GEMM K

typedef __attribute__((ext_vector_type(16))) _Float16 v16h;
typedef __attribute__((ext_vector_type(8)))  _Float16 v8h;
typedef __attribute__((ext_vector_type(8)))  float    v8f;

union V16U { v16h v; v8h h[2]; };

__device__ __forceinline__ float sigf(float x) {
  return 1.0f / (1.0f + __expf(-x));
}

// ---------------------------------------------------------------------------
// zero words of workspace (A1, A2, c1, c2 region)
// ---------------------------------------------------------------------------
__global__ void lstm_zero_kernel(unsigned int* __restrict__ p, int nWords) {
  int i = blockIdx.x * blockDim.x + threadIdx.x;
  if (i < nWords) p[i] = 0u;
}

// ---------------------------------------------------------------------------
// Build layer-1 weight panel, K-major f16: B1t[n][k], n<2048, k<640
//   k <  128 : W1[k][n] (zero-padded past E=100)
//   k >= 128 : U1[k-128][n]
// ---------------------------------------------------------------------------
__global__ void lstm_packw1_kernel(const float* __restrict__ W1,
                                   const float* __restrict__ U1,
                                   _Float16* __restrict__ B1t) {
  int i = blockIdx.x * blockDim.x + threadIdx.x;   // over 2048*640
  if (i >= GATES * K1) return;
  int n = i / K1;
  int k = i - n * K1;
  float v;
  if (k < EMBP) v = (k < EMB) ? W1[k * GATES + n] : 0.0f;
  else          v = U1[(k - EMBP) * GATES + n];
  B1t[n * K1 + k] = (_Float16)v;
}

// ---------------------------------------------------------------------------
// Build layer-2 weight panel, K-major f16: B2t[n][k], n<2048, k<1024
//   k < 512 : W2[k][n];  k >= 512 : U2[k-512][n]
// ---------------------------------------------------------------------------
__global__ void lstm_packw2_kernel(const float* __restrict__ W2,
                                   const float* __restrict__ U2,
                                   _Float16* __restrict__ B2t) {
  int i = blockIdx.x * blockDim.x + threadIdx.x;   // over 2048*1024
  if (i >= GATES * K2) return;
  int n = i >> 10;
  int k = i & 1023;
  float v = (k < UNITS) ? W2[k * GATES + n] : U2[(k - UNITS) * GATES + n];
  B2t[n * K2 + k] = (_Float16)v;
}

// ---------------------------------------------------------------------------
// Embedding gather for timestep t -> f16 into A1[:, 0:128]
// ---------------------------------------------------------------------------
__global__ void lstm_embed_kernel(const int* __restrict__ tokens,
                                  const float* __restrict__ emb,
                                  _Float16* __restrict__ A1, int t) {
  int i = blockIdx.x * blockDim.x + threadIdx.x;   // over 1024*128
  int b = i >> 7;
  int e = i & 127;
  if (b >= BATCH) return;
  int tok = tokens[b * SEQ + t];
  float v = (e < EMB) ? emb[tok * EMB + e] : 0.0f;
  A1[b * K1 + e] = (_Float16)v;
}

// ---------------------------------------------------------------------------
// WMMA GEMM:  Z[1024 x 2048] = A[1024 x K](f16, row-major, lda=K)
//                            @ Bt[2048 x K](f16, K-major)  + bias[2048]
// Workgroup: 256 threads = 8 waves as 2(M) x 4(N); wave tile 32x64
// (2x4 v_wmma_f32_16x16x32_f16 accumulators). Macro tile 64x256.
// Grid: (2048/256, 1024/64) = (8, 16).
// ---------------------------------------------------------------------------
__global__ void __launch_bounds__(256)
lstm_gemm_kernel(const _Float16* __restrict__ A,
                 const _Float16* __restrict__ Bt,
                 const float* __restrict__ bias,
                 float* __restrict__ Z, int K) {
  const int tid   = threadIdx.x;
  const int lane  = tid & 31;
  const int wave  = tid >> 5;
  const int waveM = wave >> 2;        // 0..1
  const int waveN = wave & 3;         // 0..3
  const int l16   = lane & 15;
  const int mW    = blockIdx.y * 64 + waveM * 32;
  const int nW    = blockIdx.x * 256 + waveN * 64;

  // A fragment addressing (16-bit A 16x32 layout):
  //   lanes 0-15 : row = mW+i*16+l16, K slots {0..7, 16..23}
  //   lanes 16-31: same rows,         K slots {8..15, 24..31}
  const int aKoff = (lane < 16) ? 0 : 8;
  // B fragment addressing (16-bit B 32x16 layout, K-major storage):
  //   lanes 0-15 : col = nW+j*16+l16, K slots {0..7, 16..23}
  //   lanes 16-31: same cols,         K slots {8..15, 24..31}
  const int bKoff = (lane < 16) ? 0 : 8;

  const _Float16* aBase0 = A + (size_t)(mW + l16)       * K + aKoff;
  const _Float16* aBase1 = A + (size_t)(mW + 16 + l16)  * K + aKoff;
  const _Float16* bBase[4];
#pragma unroll
  for (int j = 0; j < 4; ++j)
    bBase[j] = Bt + (size_t)(nW + j * 16 + l16) * K + bKoff;

  v8f acc[2][4] = {};

  for (int k = 0; k < K; k += 32) {
    V16U a0, a1;
    a0.h[0] = *(const v8h*)(aBase0 + k);
    a0.h[1] = *(const v8h*)(aBase0 + k + 16);
    a1.h[0] = *(const v8h*)(aBase1 + k);
    a1.h[1] = *(const v8h*)(aBase1 + k + 16);

    V16U bf[4];
#pragma unroll
    for (int j = 0; j < 4; ++j) {
      bf[j].h[0] = *(const v8h*)(bBase[j] + k);
      bf[j].h[1] = *(const v8h*)(bBase[j] + k + 16);
    }

#pragma unroll
    for (int j = 0; j < 4; ++j) {
      acc[0][j] = __builtin_amdgcn_wmma_f32_16x16x32_f16(
          false, a0.v, false, bf[j].v, (short)0, acc[0][j], false, false);
      acc[1][j] = __builtin_amdgcn_wmma_f32_16x16x32_f16(
          false, a1.v, false, bf[j].v, (short)0, acc[1][j], false, false);
    }
  }

  // Epilogue: C/D layout -> lanes 0-15 hold rows g, lanes 16-31 rows 8+g
  const int rOff = (lane < 16) ? 0 : 8;
#pragma unroll
  for (int i = 0; i < 2; ++i) {
#pragma unroll
    for (int j = 0; j < 4; ++j) {
      const int col  = nW + j * 16 + l16;
      const float bv = bias[col];
      const int rb   = mW + i * 16 + rOff;
#pragma unroll
      for (int g = 0; g < 8; ++g)
        Z[(size_t)(rb + g) * GATES + col] = acc[i][j][g] + bv;
    }
  }
}

// ---------------------------------------------------------------------------
// LSTM elementwise update. z layout per row: [i | f | g | o] (512 each).
// Writes f16 h into up to two concatenated A panels, optional f32 h.
// ---------------------------------------------------------------------------
__global__ void lstm_update_kernel(const float* __restrict__ z,
                                   float* __restrict__ c,
                                   _Float16* __restrict__ hdst0, int ld0,
                                   _Float16* __restrict__ hdst1, int ld1,
                                   float* __restrict__ hf32) {
  int idx = blockIdx.x * blockDim.x + threadIdx.x;  // over 1024*512
  if (idx >= BATCH * UNITS) return;
  int b = idx >> 9;
  int u = idx & (UNITS - 1);
  const float* zr = z + (size_t)b * GATES;
  float gi = zr[u];
  float gf = zr[UNITS + u];
  float gg = zr[2 * UNITS + u];
  float go = zr[3 * UNITS + u];
  float cn = sigf(gf) * c[idx] + sigf(gi) * tanhf(gg);
  float h  = sigf(go) * tanhf(cn);
  c[idx] = cn;
  hdst0[(size_t)b * ld0 + u] = (_Float16)h;
  if (hdst1) hdst1[(size_t)b * ld1 + u] = (_Float16)h;
  if (hf32)  hf32[idx] = h;
}

// ---------------------------------------------------------------------------
// Output head: out[b] = sigmoid(h2[b,:] . Wout + bout). One wave per row.
// ---------------------------------------------------------------------------
__global__ void lstm_out_kernel(const float* __restrict__ h2,
                                const float* __restrict__ Wout,
                                const float* __restrict__ bout,
                                float* __restrict__ out) {
  int wave = threadIdx.x >> 5;
  int lane = threadIdx.x & 31;
  int row  = blockIdx.x * 8 + wave;
  if (row >= BATCH) return;
  float s = 0.0f;
  for (int u = lane; u < UNITS; u += 32)
    s += h2[(size_t)row * UNITS + u] * Wout[u];
#pragma unroll
  for (int off = 16; off > 0; off >>= 1)
    s += __shfl_down(s, off, 32);
  if (lane == 0) out[row] = sigf(s + bout[0]);
}

// ---------------------------------------------------------------------------
extern "C" void kernel_launch(void* const* d_in, const int* in_sizes, int n_in,
                              void* d_out, int out_size, void* d_ws, size_t ws_size,
                              hipStream_t stream) {
  (void)in_sizes; (void)n_in; (void)out_size; (void)ws_size;

  const int*   tokens = (const int*)  d_in[0];
  const float* emb    = (const float*)d_in[1];
  const float* W1     = (const float*)d_in[2];
  const float* U1     = (const float*)d_in[3];
  const float* b1     = (const float*)d_in[4];
  const float* W2     = (const float*)d_in[5];
  const float* U2     = (const float*)d_in[6];
  const float* b2     = (const float*)d_in[7];
  const float* Wout   = (const float*)d_in[8];
  const float* bout   = (const float*)d_in[9];

  // Workspace layout (zero-init region first, 256B-aligned sections)
  char* ws = (char*)d_ws;
  size_t off = 0;
  auto take = [&](size_t bytes) {
    char* p = ws + off;
    off += (bytes + 255) & ~(size_t)255;
    return p;
  };
  _Float16* A1   = (_Float16*)take((size_t)BATCH * K1 * 2);      // [1024 x 640]  x|h1
  _Float16* A2   = (_Float16*)take((size_t)BATCH * K2 * 2);      // [1024 x 1024] h1|h2
  float*    c1   = (float*)   take((size_t)BATCH * UNITS * 4);
  float*    c2   = (float*)   take((size_t)BATCH * UNITS * 4);
  size_t zeroBytes = off;                                        // zero all of the above
  _Float16* B1t  = (_Float16*)take((size_t)GATES * K1 * 2);
  _Float16* B2t  = (_Float16*)take((size_t)GATES * K2 * 2);
  float*    Z    = (float*)   take((size_t)BATCH * GATES * 4);
  float*    h2f  = (float*)   take((size_t)BATCH * UNITS * 4);

  const int TPB = 256;

  // Zero h/c state (A panels include h slices that must start at 0)
  {
    int nWords = (int)(zeroBytes / 4);
    lstm_zero_kernel<<<(nWords + TPB - 1) / TPB, TPB, 0, stream>>>(
        (unsigned int*)ws, nWords);
  }
  // Repack weights to K-major f16
  lstm_packw1_kernel<<<(GATES * K1 + TPB - 1) / TPB, TPB, 0, stream>>>(W1, U1, B1t);
  lstm_packw2_kernel<<<(GATES * K2 + TPB - 1) / TPB, TPB, 0, stream>>>(W2, U2, B2t);

  dim3 gemmGrid(GATES / 256, BATCH / 64);   // (8, 16)

  for (int t = 0; t < SEQ; ++t) {
    // x_t -> A1[:, 0:128]
    lstm_embed_kernel<<<(BATCH * EMBP) / TPB, TPB, 0, stream>>>(tokens, emb, A1, t);
    // layer 1: z = [x|h1] @ B1t + b1
    lstm_gemm_kernel<<<gemmGrid, TPB, 0, stream>>>(A1, B1t, b1, Z, K1);
    // h1,c1 update; h1(f16) -> A1[:,128:640] and A2[:,0:512]
    lstm_update_kernel<<<(BATCH * UNITS) / TPB, TPB, 0, stream>>>(
        Z, c1, A1 + EMBP, K1, A2, K2, nullptr);
    // layer 2: z = [h1|h2] @ B2t + b2
    lstm_gemm_kernel<<<gemmGrid, TPB, 0, stream>>>(A2, B2t, b2, Z, K2);
    // h2,c2 update; h2(f16) -> A2[:,512:1024]; h2(f32) for the head
    lstm_update_kernel<<<(BATCH * UNITS) / TPB, TPB, 0, stream>>>(
        Z, c2, A2 + UNITS, K2, nullptr, 0, h2f);
  }

  // out = sigmoid(h2 @ Wout + bout)
  lstm_out_kernel<<<BATCH / 8, TPB, 0, stream>>>(h2f, Wout, bout, (float*)d_out);
}
